// GlimpseNetwork_41137196761766
// MI455X (gfx1250) — matrix operand order repaired and checked by reading, not compiled, not run on hardware
//
#include <hip/hip_runtime.h>
#include <hip/hip_bf16.h>

typedef __bf16 bf16;
typedef __attribute__((ext_vector_type(16))) __bf16 v16bf_t;
typedef __attribute__((ext_vector_type(8)))  __bf16 v8bf_t;
typedef __attribute__((ext_vector_type(8)))  float  v8f_t;

#define BN_EPS 1e-5f

// ---------------------------------------------------------------------------
// Foveate: extract 44x44 patch per batch (pad 22), replicating reference clamp
// logic. Output phi bf16 [128][3][44][44].
// ---------------------------------------------------------------------------
__global__ void foveate_kernel(const float* __restrict__ x,
                               const float* __restrict__ l,
                               bf16* __restrict__ phi) {
  const int b = blockIdx.x;
  const float l0 = l[b * 2 + 0];
  const float l1 = l[b * 2 + 1];
  const int fy = (int)(0.5f * (l0 + 1.0f) * 224.0f);  // start[:,0]
  const int fx = (int)(0.5f * (l1 + 1.0f) * 224.0f);  // start[:,1]
  const int ty = fy + 44, tx = fx + 44;
  const bool exceeds = (fx < 0) || (fy < 0) || (tx > 224) || (ty > 224);
  const bool y_hi = (fy > 224) || (ty > 224);
  const bool x_hi = (fx > 224) || (tx > 224);
  int ffy = y_hi ? 180 : fy;
  int ffx = x_hi ? 180 : fx;
  if (!x_hi && fx < 0) ffx = 0;
  if (!x_hi && fx >= 0 && fy < 0) ffy = 0;
  int row0 = exceeds ? ffx : fx;
  int col0 = exceeds ? ffy : fy;
  // dynamic_slice clamp to [0, 268-44]
  row0 = row0 < 0 ? 0 : (row0 > 224 ? 224 : row0);
  col0 = col0 < 0 ? 0 : (col0 > 224 ? 224 : col0);

  for (int e = threadIdx.x; e < 3 * 44 * 44; e += blockDim.x) {
    const int j = e % 44;
    const int i = (e / 44) % 44;
    const int c = e / (44 * 44);
    const int sy = row0 + i - 22;  // index into original 224x224 image
    const int sx = col0 + j - 22;
    float v = 0.0f;
    if (sy >= 0 && sy < 224 && sx >= 0 && sx < 224)
      v = x[((size_t)(b * 3 + c) * 224 + sy) * 224 + sx];
    phi[(size_t)b * (3 * 44 * 44) + e] = (bf16)v;
  }
}

// ---------------------------------------------------------------------------
// Weight pad/convert: f32 [M][Kdim] -> bf16 [M][Kpad] (zero tail)
// ---------------------------------------------------------------------------
__global__ void wpad_kernel(const float* __restrict__ w, bf16* __restrict__ wp,
                            int Kdim, int Kpad, int total) {
  const int i = blockIdx.x * blockDim.x + threadIdx.x;
  if (i >= total) return;
  const int m = i / Kpad, k = i % Kpad;
  wp[i] = (k < Kdim) ? (bf16)w[(size_t)m * Kdim + k] : (bf16)0.0f;
}

// ---------------------------------------------------------------------------
// Generic im2col: act bf16 [B][Cin][IH][IW] -> col bf16 [B*OH*OW][Kpad]
// k = c*KH*KW + kh*KW + kw ; zero for OOB and k >= Kdim
// ---------------------------------------------------------------------------
__global__ void im2col_kernel(const bf16* __restrict__ in, bf16* __restrict__ col,
                              int B, int Cin, int IH, int IW, int KH, int KW,
                              int stride, int pad, int OH, int OW,
                              int Kdim, int Kpad) {
  const int idx = blockIdx.x * blockDim.x + threadIdx.x;
  const int total = B * OH * OW * Kpad;
  if (idx >= total) return;
  const int k = idx % Kpad;
  const int n = idx / Kpad;
  bf16 v = (bf16)0.0f;
  if (k < Kdim) {
    const int c = k / (KH * KW);
    const int r = k % (KH * KW);
    const int kh = r / KW, kw = r % KW;
    const int pix = n % (OH * OW);
    const int b = n / (OH * OW);
    const int oy = pix / OW, ox = pix % OW;
    const int iy = oy * stride - pad + kh;
    const int ix = ox * stride - pad + kw;
    if (iy >= 0 && iy < IH && ix >= 0 && ix < IW)
      v = in[((size_t)(b * Cin + c) * IH + iy) * IW + ix];
  }
  col[idx] = v;
}

// ---------------------------------------------------------------------------
// WMMA implicit-GEMM conv + fused BN epilogue.
//   D[M,N] = Wp[M,Kpad] x Col[N,Kpad]^T   (Col rows are im2col columns)
// Each wave computes a 16M x 64N macro-tile: the A fragment (weights) is
// loaded once per K-step and reused across 4 v_wmma_f32_16x16x32_bf16 issues
// (4 f32 accumulators), so the load:MMA ratio drops to 10:4 b128s per 4 MMAs.
// mode 0: relu(bn(acc))          -> out bf16 NCHW
// mode 1: relu(bn(acc)+res)      -> out bf16 NCHW (res bf16 same layout)
// mode 2: bn(acc)                -> out bf16 NCHW (downsample branch)
// ---------------------------------------------------------------------------
__global__ __launch_bounds__(128) void gemm_bn_kernel(
    const bf16* __restrict__ Wp, const bf16* __restrict__ Col,
    const float* __restrict__ gn, const float* __restrict__ bt,
    const float* __restrict__ mu, const float* __restrict__ vr,
    const bf16* __restrict__ res, bf16* __restrict__ out,
    int M, int N, int Kpad, int OHW, int mode) {
  const int wave = threadIdx.x >> 5;
  const int lane = threadIdx.x & 31;
  const int mt = M >> 4;
  const int tiles = mt * (N >> 6);      // N is always a multiple of 64
  const int t = blockIdx.x * 4 + wave;
  if (t >= tiles) return;  // wave-uniform: surviving waves keep full EXEC
  const int mtile = t % mt;
  const int ntile = t / mt;
  const int half = lane >> 4;
  const int lan16 = lane & 15;

  // A-frag row: M = mtile*16 + (lane&15); 16 consecutive K at 16h
  const bf16* arow = Wp + (size_t)(mtile * 16 + lan16) * Kpad + (half << 4);
  // B-frag rows: N = ntile*64 + 16*s + (lane&15); 16 consecutive K at 16h
  const bf16* brow = Col + (size_t)(ntile * 64 + lan16) * Kpad + (half << 4);
  const size_t bstride = (size_t)16 * Kpad;  // between n-subtiles

  v8f_t acc0 = {}, acc1 = {}, acc2 = {}, acc3 = {};
  for (int k = 0; k < Kpad; k += 32) {
    v8bf_t a0 = *(const v8bf_t*)(arow + k);
    v8bf_t a1 = *(const v8bf_t*)(arow + k + 8);
    v16bf_t A;
#pragma unroll
    for (int i = 0; i < 8; ++i) { A[i] = a0[i]; A[i + 8] = a1[i]; }

#pragma unroll
    for (int s = 0; s < 4; ++s) {
      const bf16* bp = brow + bstride * s + k;
      v8bf_t b0 = *(const v8bf_t*)(bp);
      v8bf_t b1 = *(const v8bf_t*)(bp + 8);
      v16bf_t Bm;
#pragma unroll
      for (int i = 0; i < 8; ++i) { Bm[i] = b0[i]; Bm[i + 8] = b1[i]; }
      if (s == 0)
        acc0 = __builtin_amdgcn_wmma_f32_16x16x32_bf16(
            false, A, false, Bm, (short)0, acc0, false, false);
      else if (s == 1)
        acc1 = __builtin_amdgcn_wmma_f32_16x16x32_bf16(
            false, A, false, Bm, (short)0, acc1, false, false);
      else if (s == 2)
        acc2 = __builtin_amdgcn_wmma_f32_16x16x32_bf16(
            false, A, false, Bm, (short)0, acc2, false, false);
      else
        acc3 = __builtin_amdgcn_wmma_f32_16x16x32_bf16(
            false, A, false, Bm, (short)0, acc3, false, false);
    }
  }

  // Epilogue. C/D layout: VGPR r -> M = r + 8*half, N-lane = lane&15
#pragma unroll
  for (int s = 0; s < 4; ++s) {
    const v8f_t acc = (s == 0) ? acc0 : (s == 1) ? acc1 : (s == 2) ? acc2 : acc3;
    const int n = ntile * 64 + s * 16 + lan16;
    const int bidx = n / OHW;
    const int pix = n % OHW;
#pragma unroll
    for (int r = 0; r < 8; ++r) {
      const int m = mtile * 16 + r + half * 8;
      const float scale = gn[m] * __frsqrt_rn(vr[m] + BN_EPS);
      float val = (acc[r] - mu[m]) * scale + bt[m];
      const size_t oidx = ((size_t)bidx * M + m) * OHW + pix;
      if (mode == 1) val += (float)res[oidx];
      if (mode != 2) val = fmaxf(val, 0.0f);
      out[oidx] = (bf16)val;
    }
  }
}

// ---------------------------------------------------------------------------
// Maxpool 3x3 s2 p1 : bf16 [128][64][22][22] -> bf16 [128][64][11][11]
// ---------------------------------------------------------------------------
__global__ void maxpool_kernel(const bf16* __restrict__ in, bf16* __restrict__ out) {
  const int i = blockIdx.x * blockDim.x + threadIdx.x;
  if (i >= 128 * 64 * 11 * 11) return;
  const int ox = i % 11;
  const int oy = (i / 11) % 11;
  const int bc = i / 121;
  float mx = -3.402823466e38f;
#pragma unroll
  for (int dy = 0; dy < 3; ++dy) {
#pragma unroll
    for (int dx = 0; dx < 3; ++dx) {
      const int iy = 2 * oy - 1 + dy;
      const int ix = 2 * ox - 1 + dx;
      if (iy >= 0 && iy < 22 && ix >= 0 && ix < 22)
        mx = fmaxf(mx, (float)in[(size_t)bc * 484 + iy * 22 + ix]);
    }
  }
  out[i] = (bf16)mx;
}

// Global average pool: bf16 [128][C][HW] -> f32 [128][C]
__global__ void avgpool_kernel(const bf16* __restrict__ in, float* __restrict__ out,
                               int C, int HW) {
  const int i = blockIdx.x * blockDim.x + threadIdx.x;
  if (i >= 128 * C) return;
  float s = 0.0f;
  for (int j = 0; j < HW; ++j) s += (float)in[(size_t)i * HW + j];
  out[i] = s / (float)HW;
}

// Dense: out[b][o] = (relu?) (bias[o] + dot(in[b], w[o]))  ; w is [On][In]
__global__ void fc_kernel(const float* __restrict__ in, const float* __restrict__ w,
                          const float* __restrict__ bias, float* __restrict__ out,
                          int Bn, int In, int On, int do_relu) {
  const int i = blockIdx.x * blockDim.x + threadIdx.x;
  if (i >= Bn * On) return;
  const int b = i / On, o = i % On;
  float s = bias[o];
  for (int k = 0; k < In; ++k) s += in[b * In + k] * w[o * In + k];
  out[i] = do_relu ? fmaxf(s, 0.0f) : s;
}

// out[b][o] = relu(b3[o] + b4[o] + dot(phi,W3[o]) + dot(l,W4[o]))
__global__ void fc_final_kernel(const float* __restrict__ phi_out,
                                const float* __restrict__ l_out,
                                const float* __restrict__ w3, const float* __restrict__ b3,
                                const float* __restrict__ w4, const float* __restrict__ b4,
                                float* __restrict__ out) {
  const int i = blockIdx.x * blockDim.x + threadIdx.x;
  if (i >= 128 * 256) return;
  const int b = i / 256, o = i % 256;
  float s = b3[o] + b4[o];
  for (int k = 0; k < 128; ++k)
    s += phi_out[b * 128 + k] * w3[o * 128 + k] + l_out[b * 128 + k] * w4[o * 128 + k];
  out[i] = fmaxf(s, 0.0f);
}

// ---------------------------------------------------------------------------
// Host-side orchestration
// ---------------------------------------------------------------------------
// Input index map (x, l, then params flattened in JAX pytree order:
// sorted dict keys, tuples in order, lists in order):
enum {
  IN_X = 0, IN_L = 1,
  IN_BN1_G = 2, IN_BN1_B, IN_BN1_M, IN_BN1_V,
  IN_CONV1_W = 6,
  IN_FC1_B = 7, IN_FC1_W, IN_FC2_B, IN_FC2_W, IN_FC3_B, IN_FC3_W, IN_FC4_B, IN_FC4_W,
  IN_L1B0_BN1_G = 15, IN_L1B0_BN1_B, IN_L1B0_BN1_M, IN_L1B0_BN1_V,
  IN_L1B0_BN2_G = 19, IN_L1B0_BN2_B, IN_L1B0_BN2_M, IN_L1B0_BN2_V,
  IN_L1B0_C1 = 23, IN_L1B0_C2 = 24,
  IN_L1B1_BN1_G = 25, IN_L1B1_BN1_B, IN_L1B1_BN1_M, IN_L1B1_BN1_V,
  IN_L1B1_BN2_G = 29, IN_L1B1_BN2_B, IN_L1B1_BN2_M, IN_L1B1_BN2_V,
  IN_L1B1_C1 = 33, IN_L1B1_C2 = 34,
  IN_L2B0_BN1_G = 35, IN_L2B0_BN1_B, IN_L2B0_BN1_M, IN_L2B0_BN1_V,
  IN_L2B0_BN2_G = 39, IN_L2B0_BN2_B, IN_L2B0_BN2_M, IN_L2B0_BN2_V,
  IN_L2B0_C1 = 43, IN_L2B0_C2 = 44,
  IN_L2B0_DBN_G = 45, IN_L2B0_DBN_B, IN_L2B0_DBN_M, IN_L2B0_DBN_V,
  IN_L2B0_DC = 49,
  IN_L2B1_BN1_G = 50, IN_L2B1_BN1_B, IN_L2B1_BN1_M, IN_L2B1_BN1_V,
  IN_L2B1_BN2_G = 54, IN_L2B1_BN2_B, IN_L2B1_BN2_M, IN_L2B1_BN2_V,
  IN_L2B1_C1 = 58, IN_L2B1_C2 = 59
};

static inline void* bump(char* base, size_t& off, size_t bytes) {
  off = (off + 255) & ~(size_t)255;
  void* p = base + off;
  off += bytes;
  return p;
}

static void run_gemm(hipStream_t stream, const bf16* Wp, const bf16* Col,
                     const float* g, const float* b, const float* m, const float* v,
                     const bf16* res, bf16* out, int M, int N, int Kpad, int OHW,
                     int mode) {
  const int tiles = (M / 16) * (N / 64);
  const int blocks = (tiles + 3) / 4;
  gemm_bn_kernel<<<blocks, 128, 0, stream>>>(Wp, Col, g, b, m, v, res, out,
                                             M, N, Kpad, OHW, mode);
}

static void run_im2col(hipStream_t stream, const bf16* in, bf16* col, int B, int Cin,
                       int IH, int IW, int KH, int KW, int stride, int pad,
                       int OH, int OW, int Kdim, int Kpad) {
  const int total = B * OH * OW * Kpad;
  im2col_kernel<<<(total + 255) / 256, 256, 0, stream>>>(
      in, col, B, Cin, IH, IW, KH, KW, stride, pad, OH, OW, Kdim, Kpad);
}

extern "C" void kernel_launch(void* const* d_in, const int* in_sizes, int n_in,
                              void* d_out, int out_size, void* d_ws, size_t ws_size,
                              hipStream_t stream) {
  (void)in_sizes; (void)n_in; (void)out_size; (void)ws_size;
  const float* x = (const float*)d_in[IN_X];
  const float* l = (const float*)d_in[IN_L];
  auto F = [&](int i) { return (const float*)d_in[i]; };

  char* base = (char*)d_ws;
  size_t off = 0;
  // activations
  bf16* phi = (bf16*)bump(base, off, (size_t)128 * 3 * 44 * 44 * 2);
  bf16* h1  = (bf16*)bump(base, off, (size_t)128 * 64 * 22 * 22 * 2);
  bf16* p1  = (bf16*)bump(base, off, (size_t)128 * 64 * 11 * 11 * 2);
  bf16* a11 = (bf16*)bump(base, off, (size_t)128 * 64 * 11 * 11 * 2);
  bf16* o11 = (bf16*)bump(base, off, (size_t)128 * 64 * 11 * 11 * 2);
  bf16* a12 = (bf16*)bump(base, off, (size_t)128 * 64 * 11 * 11 * 2);
  bf16* o12 = (bf16*)bump(base, off, (size_t)128 * 64 * 11 * 11 * 2);
  bf16* d2  = (bf16*)bump(base, off, (size_t)128 * 128 * 6 * 6 * 2);
  bf16* a21 = (bf16*)bump(base, off, (size_t)128 * 128 * 6 * 6 * 2);
  bf16* o21 = (bf16*)bump(base, off, (size_t)128 * 128 * 6 * 6 * 2);
  bf16* a22 = (bf16*)bump(base, off, (size_t)128 * 128 * 6 * 6 * 2);
  bf16* o22 = (bf16*)bump(base, off, (size_t)128 * 128 * 6 * 6 * 2);
  // shared im2col buffer (max: conv1, 61952*160)
  bf16* col = (bf16*)bump(base, off, (size_t)61952 * 160 * 2);
  // padded bf16 weights
  bf16* wpc1   = (bf16*)bump(base, off, (size_t)64 * 160 * 2);
  bf16* wp10c1 = (bf16*)bump(base, off, (size_t)64 * 576 * 2);
  bf16* wp10c2 = (bf16*)bump(base, off, (size_t)64 * 576 * 2);
  bf16* wp11c1 = (bf16*)bump(base, off, (size_t)64 * 576 * 2);
  bf16* wp11c2 = (bf16*)bump(base, off, (size_t)64 * 576 * 2);
  bf16* wp20c1 = (bf16*)bump(base, off, (size_t)128 * 576 * 2);
  bf16* wp20c2 = (bf16*)bump(base, off, (size_t)128 * 1152 * 2);
  bf16* wp20d  = (bf16*)bump(base, off, (size_t)128 * 64 * 2);
  bf16* wp21c1 = (bf16*)bump(base, off, (size_t)128 * 1152 * 2);
  bf16* wp21c2 = (bf16*)bump(base, off, (size_t)128 * 1152 * 2);
  // head
  float* feat    = (float*)bump(base, off, (size_t)128 * 128 * 4);
  float* phi_out = (float*)bump(base, off, (size_t)128 * 128 * 4);
  float* l_out   = (float*)bump(base, off, (size_t)128 * 128 * 4);

  // --- convert weights to padded bf16 ---
  auto wpad = [&](const float* w, bf16* wp, int M, int Kdim, int Kpad) {
    const int total = M * Kpad;
    wpad_kernel<<<(total + 255) / 256, 256, 0, stream>>>(w, wp, Kdim, Kpad, total);
  };
  wpad(F(IN_CONV1_W), wpc1, 64, 147, 160);
  wpad(F(IN_L1B0_C1), wp10c1, 64, 576, 576);
  wpad(F(IN_L1B0_C2), wp10c2, 64, 576, 576);
  wpad(F(IN_L1B1_C1), wp11c1, 64, 576, 576);
  wpad(F(IN_L1B1_C2), wp11c2, 64, 576, 576);
  wpad(F(IN_L2B0_C1), wp20c1, 128, 576, 576);
  wpad(F(IN_L2B0_C2), wp20c2, 128, 1152, 1152);
  wpad(F(IN_L2B0_DC), wp20d, 128, 64, 64);
  wpad(F(IN_L2B1_C1), wp21c1, 128, 1152, 1152);
  wpad(F(IN_L2B1_C2), wp21c2, 128, 1152, 1152);

  // --- foveate ---
  foveate_kernel<<<128, 256, 0, stream>>>(x, l, phi);

  // --- conv1 (7x7 s2 p3) + bn1 + relu ---
  run_im2col(stream, phi, col, 128, 3, 44, 44, 7, 7, 2, 3, 22, 22, 147, 160);
  run_gemm(stream, wpc1, col, F(IN_BN1_G), F(IN_BN1_B), F(IN_BN1_M), F(IN_BN1_V),
           nullptr, h1, 64, 128 * 484, 160, 484, 0);

  // --- maxpool 3x3 s2 p1 ---
  maxpool_kernel<<<(128 * 64 * 121 + 255) / 256, 256, 0, stream>>>(h1, p1);

  // --- layer1 block0 ---
  run_im2col(stream, p1, col, 128, 64, 11, 11, 3, 3, 1, 1, 11, 11, 576, 576);
  run_gemm(stream, wp10c1, col, F(IN_L1B0_BN1_G), F(IN_L1B0_BN1_B),
           F(IN_L1B0_BN1_M), F(IN_L1B0_BN1_V), nullptr, a11,
           64, 128 * 121, 576, 121, 0);
  run_im2col(stream, a11, col, 128, 64, 11, 11, 3, 3, 1, 1, 11, 11, 576, 576);
  run_gemm(stream, wp10c2, col, F(IN_L1B0_BN2_G), F(IN_L1B0_BN2_B),
           F(IN_L1B0_BN2_M), F(IN_L1B0_BN2_V), p1, o11,
           64, 128 * 121, 576, 121, 1);

  // --- layer1 block1 ---
  run_im2col(stream, o11, col, 128, 64, 11, 11, 3, 3, 1, 1, 11, 11, 576, 576);
  run_gemm(stream, wp11c1, col, F(IN_L1B1_BN1_G), F(IN_L1B1_BN1_B),
           F(IN_L1B1_BN1_M), F(IN_L1B1_BN1_V), nullptr, a12,
           64, 128 * 121, 576, 121, 0);
  run_im2col(stream, a12, col, 128, 64, 11, 11, 3, 3, 1, 1, 11, 11, 576, 576);
  run_gemm(stream, wp11c2, col, F(IN_L1B1_BN2_G), F(IN_L1B1_BN2_B),
           F(IN_L1B1_BN2_M), F(IN_L1B1_BN2_V), o11, o12,
           64, 128 * 121, 576, 121, 1);

  // --- layer2 block0 (stride 2, downsample) ---
  // downsample 1x1 s2 -> bn (no relu)
  run_im2col(stream, o12, col, 128, 64, 11, 11, 1, 1, 2, 0, 6, 6, 64, 64);
  run_gemm(stream, wp20d, col, F(IN_L2B0_DBN_G), F(IN_L2B0_DBN_B),
           F(IN_L2B0_DBN_M), F(IN_L2B0_DBN_V), nullptr, d2,
           128, 128 * 36, 64, 36, 2);
  // c1 3x3 s2 p1 + bn + relu
  run_im2col(stream, o12, col, 128, 64, 11, 11, 3, 3, 2, 1, 6, 6, 576, 576);
  run_gemm(stream, wp20c1, col, F(IN_L2B0_BN1_G), F(IN_L2B0_BN1_B),
           F(IN_L2B0_BN1_M), F(IN_L2B0_BN1_V), nullptr, a21,
           128, 128 * 36, 576, 36, 0);
  // c2 3x3 s1 p1 + bn + add(d2) + relu
  run_im2col(stream, a21, col, 128, 128, 6, 6, 3, 3, 1, 1, 6, 6, 1152, 1152);
  run_gemm(stream, wp20c2, col, F(IN_L2B0_BN2_G), F(IN_L2B0_BN2_B),
           F(IN_L2B0_BN2_M), F(IN_L2B0_BN2_V), d2, o21,
           128, 128 * 36, 1152, 36, 1);

  // --- layer2 block1 ---
  run_im2col(stream, o21, col, 128, 128, 6, 6, 3, 3, 1, 1, 6, 6, 1152, 1152);
  run_gemm(stream, wp21c1, col, F(IN_L2B1_BN1_G), F(IN_L2B1_BN1_B),
           F(IN_L2B1_BN1_M), F(IN_L2B1_BN1_V), nullptr, a22,
           128, 128 * 36, 1152, 36, 0);
  run_im2col(stream, a22, col, 128, 128, 6, 6, 3, 3, 1, 1, 6, 6, 1152, 1152);
  run_gemm(stream, wp21c2, col, F(IN_L2B1_BN2_G), F(IN_L2B1_BN2_B),
           F(IN_L2B1_BN2_M), F(IN_L2B1_BN2_V), o21, o22,
           128, 128 * 36, 1152, 36, 1);

  // --- head ---
  avgpool_kernel<<<(128 * 128 + 255) / 256, 256, 0, stream>>>(o22, feat, 128, 36);
  fc_kernel<<<(128 * 128 + 255) / 256, 256, 0, stream>>>(
      feat, F(IN_FC1_W), F(IN_FC1_B), phi_out, 128, 128, 128, 1);
  fc_kernel<<<(128 * 128 + 255) / 256, 256, 0, stream>>>(
      l, F(IN_FC2_W), F(IN_FC2_B), l_out, 128, 2, 128, 1);
  fc_final_kernel<<<(128 * 256 + 255) / 256, 256, 0, stream>>>(
      phi_out, l_out, F(IN_FC3_W), F(IN_FC3_B), F(IN_FC4_W), F(IN_FC4_B),
      (float*)d_out);
}
